// AttentionBlock_77592879169588
// MI455X (gfx1250) — matrix operand order repaired and checked by reading, not compiled, not run on hardware
//
#include <hip/hip_runtime.h>
#include <hip/hip_bf16.h>

#define Bn     8
#define Cn     512
#define Tn     1024
#define HEADS  8
#define CH     64          // Cn / HEADS
#define GROUPS 32
#define CPG    16          // Cn / GROUPS
#define QK_SCALE 0.3535533905932738f   // 64^-0.25

typedef __attribute__((ext_vector_type(16))) __bf16 v16bf;
typedef __attribute__((ext_vector_type(8)))  float  v8f;
typedef unsigned int   u32;
typedef unsigned short u16;

union Frag16 {
    uint4  q[2];
    u16    h[16];
    v16bf  v;
};

__device__ __forceinline__ u16 f2bf(float f) {
    u32 u = __builtin_bit_cast(u32, f);
    u32 r = (u + 0x7FFFu + ((u >> 16) & 1u)) >> 16;  // RNE
    return (u16)r;
}

__device__ __forceinline__ v8f wmma_bf16(const Frag16& a, const Frag16& b, v8f c) {
    return __builtin_amdgcn_wmma_f32_16x16x32_bf16(false, a.v, false, b.v,
                                                   (short)0, c, false, false);
}

// ---------------------------------------------------------------- weights->bf16
__global__ void conv_weights_kernel(const float* __restrict__ qkv_w,
                                    const float* __restrict__ proj_w,
                                    u16* __restrict__ wq, u16* __restrict__ wp) {
    const int nq = 3 * Cn * Cn, np = Cn * Cn;
    for (int i = blockIdx.x * blockDim.x + threadIdx.x; i < nq + np;
         i += gridDim.x * blockDim.x) {
        if (i < nq) wq[i] = f2bf(qkv_w[i]);
        else        wp[i - nq] = f2bf(proj_w[i - nq]);
    }
}

// ------------------------------------------------------- GroupNorm -> xnT (t-major)
__global__ __launch_bounds__(256) void groupnorm_kernel(
        const float* __restrict__ x, const float* __restrict__ gamma,
        const float* __restrict__ beta, u16* __restrict__ xnT) {
    const int b = blockIdx.x / GROUPS, g = blockIdx.x % GROUPS;
    const float* xp = x + ((size_t)b * Cn + g * CPG) * Tn;
    __shared__ float s1[256], s2[256];
    float sum = 0.f, sq = 0.f;
    for (int i = threadIdx.x; i < CPG * Tn; i += 256) {
        float v = xp[i]; sum += v; sq += v * v;
    }
    s1[threadIdx.x] = sum; s2[threadIdx.x] = sq;
    __syncthreads();
    for (int st = 128; st > 0; st >>= 1) {
        if (threadIdx.x < st) {
            s1[threadIdx.x] += s1[threadIdx.x + st];
            s2[threadIdx.x] += s2[threadIdx.x + st];
        }
        __syncthreads();
    }
    const float inv_n = 1.0f / (CPG * Tn);
    const float mean = s1[0] * inv_n;
    const float var  = s2[0] * inv_n - mean * mean;
    const float rstd = rsqrtf(var + 1e-5f);
    float ga[CPG], be[CPG];
    for (int c = 0; c < CPG; ++c) {
        ga[c] = gamma[g * CPG + c] * rstd;
        be[c] = beta[g * CPG + c] - mean * ga[c];
    }
    for (int t = threadIdx.x; t < Tn; t += 256) {
        __align__(16) u16 o[CPG];
        for (int c = 0; c < CPG; ++c)
            o[c] = f2bf(xp[(size_t)c * Tn + t] * ga[c] + be[c]);
        u16* dst = xnT + ((size_t)b * Tn + t) * Cn + g * CPG;
        ((uint4*)dst)[0] = *(const uint4*)&o[0];
        ((uint4*)dst)[1] = *(const uint4*)&o[8];
    }
}

// ------------------------------------------------------------- QKV WMMA GEMM
// qkv[o,t] = sum_c W[o,c] * xn[c,t];  A = W (row-major), B = xnT (t-major).
// One wave computes a 16(M) x 64(N) strip.
__global__ __launch_bounds__(256) void qkv_gemm_kernel(
        const u16* __restrict__ wq, const u16* __restrict__ xnT,
        const float* __restrict__ qkv_b,
        u16* __restrict__ qbuf, u16* __restrict__ kbuf, u16* __restrict__ vbuf) {
    const int wid  = blockIdx.x * 8 + (threadIdx.x >> 5);
    const int lane = threadIdx.x & 31;
    const int half = lane >> 4, l16 = lane & 15;
    const int b   = wid / 1536;          // 96 m-tiles * 16 n-groups
    const int rem = wid % 1536;
    const int m0  = (rem / 16) * 16;
    const int n0  = (rem % 16) * 64;

    const u16* Arow = wq + (size_t)(m0 + l16) * Cn;
    const u16* Bp   = xnT + (size_t)b * Tn * Cn;

    v8f acc[4];
    for (int i = 0; i < 4; ++i) acc[i] = (v8f)0.0f;

    for (int k0 = 0; k0 < Cn; k0 += 32) {
        Frag16 af;
        af.q[0] = *(const uint4*)(Arow + k0 + half * 16);
        af.q[1] = *(const uint4*)(Arow + k0 + half * 16 + 8);
        for (int nt = 0; nt < 4; ++nt) {
            const u16* brow = Bp + (size_t)(n0 + nt * 16 + l16) * Cn + k0 + half * 16;
            Frag16 bf;
            bf.q[0] = ((const uint4*)brow)[0];
            bf.q[1] = ((const uint4*)brow)[1];
            acc[nt] = wmma_bf16(af, bf, acc[nt]);
        }
    }
    // epilogue: bias, q/k scaling, scatter to per-head buffers
    const int head = m0 / 192;
    const int sec  = (m0 % 192) / 64;    // 0=q 1=k 2=v (64 is a multiple of 16)
    const int rb   = (m0 % 192) % 64;
    const float sc = (sec < 2) ? QK_SCALE : 1.0f;
    const int bh   = b * HEADS + head;
    for (int nt = 0; nt < 4; ++nt) {
        const int t = n0 + nt * 16 + l16;
        for (int r = 0; r < 8; ++r) {
            const int mr = half * 8 + r;
            const float val = (acc[nt][r] + qkv_b[m0 + mr]) * sc;
            const int cl = rb + mr;
            if (sec == 0)      qbuf[((size_t)bh * Tn + t) * CH + cl] = f2bf(val);
            else if (sec == 1) kbuf[((size_t)bh * Tn + t) * CH + cl] = f2bf(val);
            else               vbuf[((size_t)bh * CH + cl) * Tn + t] = f2bf(val);
        }
    }
}

// -------------------------------------------------- flash attention (one wave per
// 16-row t-tile). S^T = k^T q puts scores in exactly the A-fragment layout for P*V.
__global__ __launch_bounds__(256) void attn_kernel(
        const u16* __restrict__ qbuf, const u16* __restrict__ kbuf,
        const u16* __restrict__ vbuf, u16* __restrict__ aT) {
    const int wid  = blockIdx.x * 8 + (threadIdx.x >> 5);
    const int lane = threadIdx.x & 31;
    const int half = lane >> 4, l16 = lane & 15;
    const int bh = wid >> 6;
    const int t0 = (wid & 63) * 16;

    const u16* qb = qbuf + (size_t)bh * Tn * CH;
    const u16* kb = kbuf + (size_t)bh * Tn * CH;
    const u16* vb = vbuf + (size_t)bh * CH * Tn;

    // q B-fragments for c-chunks 0..31, 32..63 (held for whole s sweep)
    Frag16 qf[2];
    {
        const u16* row = qb + (size_t)(t0 + l16) * CH;
        for (int g = 0; g < 2; ++g) {
            const u16* p = row + g * 32 + half * 8;
            qf[g].q[0] = ((const uint4*)p)[0];
            qf[g].q[1] = *(const uint4*)(p + 16);
        }
    }

    v8f acc[4];
    for (int i = 0; i < 4; ++i) acc[i] = (v8f)0.0f;
    float m_run = -1e30f, l_run = 0.0f;

    for (int s0 = 0; s0 < Tn; s0 += 32) {
        // S^T tiles: D[m=s_local][n=t] = sum_c k[s][c] * q[t][c]
        v8f st[2];
        for (int stile = 0; stile < 2; ++stile) {
            const u16* krow = kb + (size_t)(s0 + stile * 16 + l16) * CH;
            Frag16 ka;
            v8f z = (v8f)0.0f;
            ka.q[0] = *(const uint4*)(krow + half * 8);
            ka.q[1] = *(const uint4*)(krow + 16 + half * 8);
            z = wmma_bf16(ka, qf[0], z);
            ka.q[0] = *(const uint4*)(krow + 32 + half * 8);
            ka.q[1] = *(const uint4*)(krow + 48 + half * 8);
            z = wmma_bf16(ka, qf[1], z);
            st[stile] = z;
        }
        // online softmax stats; lane's row is t = l16 (halves cover different s)
        float mx = st[0][0];
        for (int r = 1; r < 8; ++r) mx = fmaxf(mx, st[0][r]);
        for (int r = 0; r < 8; ++r) mx = fmaxf(mx, st[1][r]);
        mx = fmaxf(mx, __shfl_xor(mx, 16, 32));
        const float m_new = fmaxf(m_run, mx);
        const float alpha = __expf(m_run - m_new);
        Frag16 pf;
        float psum = 0.0f;
        for (int r = 0; r < 8; ++r) {          // in-lane S^T -> P A-fragment
            float p0 = __expf(st[0][r] - m_new);
            float p1 = __expf(st[1][r] - m_new);
            psum += p0 + p1;
            pf.h[r]     = f2bf(p0);
            pf.h[8 + r] = f2bf(p1);
        }
        psum += __shfl_xor(psum, 16, 32);
        l_run = l_run * alpha + psum;
        m_run = m_new;
        // per-register row rescale factor (row t = half*8 + r lives in lane t)
        float ar[8];
        for (int r = 0; r < 8; ++r) ar[r] = __shfl(alpha, half * 8 + r, 32);
        for (int ct = 0; ct < 4; ++ct) {
            const u16* vrow = vb + (size_t)(ct * 16 + l16) * Tn + s0 + half * 8;
            Frag16 vf;
            vf.q[0] = ((const uint4*)vrow)[0];
            vf.q[1] = *(const uint4*)(vrow + 16);
            v8f c = acc[ct];
            for (int r = 0; r < 8; ++r) c[r] *= ar[r];
            acc[ct] = wmma_bf16(pf, vf, c);
        }
    }
    // finalize: a^T[t][head*64+c], t = t0 + half*8 + r, c = ct*16 + l16
    const float linv = 1.0f / l_run;
    float lr[8];
    for (int r = 0; r < 8; ++r) lr[r] = __shfl(linv, half * 8 + r, 32);
    const int b = bh >> 3, head = bh & 7;
    for (int ct = 0; ct < 4; ++ct)
        for (int r = 0; r < 8; ++r) {
            const int t = t0 + half * 8 + r;
            const int c = head * CH + ct * 16 + l16;
            aT[((size_t)b * Tn + t) * Cn + c] = f2bf(acc[ct][r] * lr[r]);
        }
}

// ------------------------------------------------- proj WMMA GEMM + residual
__global__ __launch_bounds__(256) void proj_gemm_kernel(
        const u16* __restrict__ wp, const u16* __restrict__ aT,
        const float* __restrict__ proj_b, const float* __restrict__ x,
        float* __restrict__ out) {
    const int wid  = blockIdx.x * 8 + (threadIdx.x >> 5);
    const int lane = threadIdx.x & 31;
    const int half = lane >> 4, l16 = lane & 15;
    const int b   = wid / 512;           // 32 m-tiles * 16 n-groups
    const int rem = wid % 512;
    const int m0  = (rem / 16) * 16;
    const int n0  = (rem % 16) * 64;

    const u16* Arow = wp + (size_t)(m0 + l16) * Cn;
    const u16* Bp   = aT + (size_t)b * Tn * Cn;

    v8f acc[4];
    for (int i = 0; i < 4; ++i) acc[i] = (v8f)0.0f;

    for (int k0 = 0; k0 < Cn; k0 += 32) {
        Frag16 af;
        af.q[0] = *(const uint4*)(Arow + k0 + half * 16);
        af.q[1] = *(const uint4*)(Arow + k0 + half * 16 + 8);
        for (int nt = 0; nt < 4; ++nt) {
            const u16* brow = Bp + (size_t)(n0 + nt * 16 + l16) * Cn + k0 + half * 16;
            Frag16 bf;
            bf.q[0] = ((const uint4*)brow)[0];
            bf.q[1] = ((const uint4*)brow)[1];
            acc[nt] = wmma_bf16(af, bf, acc[nt]);
        }
    }
    for (int nt = 0; nt < 4; ++nt) {
        const int t = n0 + nt * 16 + l16;
        for (int r = 0; r < 8; ++r) {
            const int o = m0 + half * 8 + r;
            const size_t idx = ((size_t)b * Cn + o) * Tn + t;
            out[idx] = x[idx] + acc[nt][r] + proj_b[o];
        }
    }
}

extern "C" void kernel_launch(void* const* d_in, const int* in_sizes, int n_in,
                              void* d_out, int out_size, void* d_ws, size_t ws_size,
                              hipStream_t stream) {
    const float* x      = (const float*)d_in[0];
    const float* gamma  = (const float*)d_in[1];
    const float* beta   = (const float*)d_in[2];
    const float* qkv_w  = (const float*)d_in[3];
    const float* qkv_b  = (const float*)d_in[4];
    const float* proj_w = (const float*)d_in[5];
    const float* proj_b = (const float*)d_in[6];
    float* out = (float*)d_out;

    u16* wq   = (u16*)d_ws;                       // 3*512*512
    u16* wp   = wq  + (size_t)3 * Cn * Cn;        // 512*512
    u16* xnT  = wp  + (size_t)Cn * Cn;            // B*T*C
    u16* qbuf = xnT + (size_t)Bn * Tn * Cn;       // (B*H)*T*CH
    u16* kbuf = qbuf + (size_t)Bn * HEADS * Tn * CH;
    u16* vbuf = kbuf + (size_t)Bn * HEADS * Tn * CH;
    u16* aT   = vbuf + (size_t)Bn * HEADS * Tn * CH;  // B*T*C

    conv_weights_kernel<<<1024, 256, 0, stream>>>(qkv_w, proj_w, wq, wp);
    groupnorm_kernel<<<Bn * GROUPS, 256, 0, stream>>>(x, gamma, beta, xnT);
    // waves: 8 batches * 96 m-tiles * 16 n-groups = 12288 -> 1536 blocks of 8 waves
    qkv_gemm_kernel<<<1536, 256, 0, stream>>>(wq, xnT, qkv_b, qbuf, kbuf, vbuf);
    // waves: 64 bh * 64 t-tiles = 4096 -> 512 blocks
    attn_kernel<<<512, 256, 0, stream>>>(qbuf, kbuf, vbuf, aT);
    // waves: 8 * 32 * 16 = 4096 -> 512 blocks
    proj_gemm_kernel<<<512, 256, 0, stream>>>(wp, aT, proj_b, x, out);
}
